// DimeNetPP_326417514504
// MI455X (gfx1250) — matrix operand order repaired and checked by reading, not compiled, not run on hardware
//
#include <hip/hip_runtime.h>
#include <hip/hip_bf16.h>
#include <cstdint>
#include <cstddef>

// ---------------------------------------------------------------------------
// DimeNet++ forward for MI455X (gfx1250, wave32, WMMA).
// f16 activations, f32 WMMA accumulation (v_wmma_f32_16x16x32_f16).
// B weights pre-swizzled to fragment-native layout -> 1 vector load per frag.
// GEMM templated on (NSEG, SEGK, NT); gather rows resolved once per segment;
// restrict-qualified read-only operands for load hoisting / pipelining.
// ---------------------------------------------------------------------------

typedef _Float16 h16;
typedef __attribute__((ext_vector_type(8)))  _Float16 v8h;
typedef __attribute__((ext_vector_type(16))) _Float16 v16h;
typedef __attribute__((ext_vector_type(8)))  float    v8f;

#define N_NODES_C  10000
#define N_EDGES_C  150000
#define N_TRIP_C   750000
#define N_GRAPHS_C 64

// Spherical Bessel zeros z_{l,n}, l=0..6, n=0..5 (row-major l*6+n).
__constant__ float c_bz[42] = {
  3.14159265f,  6.28318531f,  9.42477796f, 12.56637061f, 15.70796327f, 18.84955592f,
  4.49340946f,  7.72525184f, 10.90412166f, 14.06619391f, 17.22075527f, 20.37130296f,
  5.76345920f,  9.09501133f, 12.32294097f, 15.51460301f, 18.68903636f, 21.85387422f,
  6.98793200f, 10.41711855f, 13.69802315f, 16.92362129f, 20.12180617f, 23.30424699f,
  8.18256145f, 11.70490715f, 15.03966471f, 18.30125596f, 21.52541773f, 24.72756555f,
  9.35581211f, 12.96653017f, 16.35470964f, 19.65315210f, 22.90455065f, 26.12775014f,
 10.51283541f, 14.20739246f, 17.64797487f, 20.98346307f, 24.26276804f, 27.50786839f
};

__device__ __forceinline__ float siluf(float x) {
  return __fdividef(x, 1.f + __expf(-x));
}

__device__ __forceinline__ float sphjn(float x, int l) {
  float s, c;
  __sincosf(x, &s, &c);
  float inv = 1.f / x;
  float j0 = s * inv;
  if (l == 0) return j0;
  float jm1 = j0;
  float jc  = (s * inv - c) * inv;
  for (int k = 2; k <= l; ++k) {
    float jn = (2.f * k - 1.f) * inv * jc - jm1;
    jm1 = jc; jc = jn;
  }
  return jc;
}

// ---------------------------------------------------------------------------
// Generic WMMA GEMM: out = epilogue( concat_s(A_s[gather_s]) @ B + bias )
//   epilogue: optional SiLU, optional elementwise f16 multiplier (row gather),
//   optional elementwise f16 addend, write f16 and/or atomic-scatter f32.
// 256 threads = 8 waves; wave w -> rows [bx*128+16w,+16),
// col tiles [by*NT, +NT). lda == SEGK by construction.
//
// A fragment (16x32 f16): lane<16 -> row=lane&15,
// K {0..15}; lane>=16 -> K+16.  Two contiguous 16B chunks per lane.
// B pre-swizzled: Bp[((kt*ntiles+nt)*32+lane)*16 + j] = B[kt*32+(lane>=16)*16+j,
// nt*16+(lane&15)] -> one 32B load per fragment.
// ---------------------------------------------------------------------------
template <int NSEG, int SEGK, int NT>
__global__ __launch_bounds__(256) void wmma_gemm(
    const h16* __restrict__ A0, const h16* __restrict__ A1,
    const h16* __restrict__ A2,
    const int* __restrict__ g0, const int* __restrict__ g1,
    const int* __restrict__ g2,
    const h16* __restrict__ Bp, int ntiles,
    const float* __restrict__ bias, int act,
    const h16* mulIn, const int* __restrict__ mulG, int mulLd,
    const h16* addIn, int addLd,
    h16* out16, int ldo,
    float* scatOut, const int* __restrict__ scatIdx, int scatLd,
    int M, int N)
{
  int wave = threadIdx.x >> 5;
  int row0 = blockIdx.x * 128 + wave * 16;
  int nt0  = blockIdx.y * NT;
  if (row0 >= M) return;                 // wave-uniform: EXEC stays all-ones
  int lane = threadIdx.x & 31;

  int r16 = row0 + (lane & 15);
  if (r16 >= M) r16 = M - 1;
  int koff = (lane >> 4) << 4;
  const h16* __restrict__ blane = Bp + (size_t)lane * 16;

  v8f acc[NT] = {};
#pragma unroll
  for (int s = 0; s < NSEG; ++s) {
    const h16* __restrict__ A = (s == 0) ? A0 : ((s == 1) ? A1 : A2);
    const int* __restrict__ g = (s == 0) ? g0 : ((s == 1) ? g1 : g2);
    int r = g ? g[r16] : r16;            // resolved ONCE per segment
    const h16* __restrict__ ap = A + (size_t)r * SEGK + koff;
#pragma unroll
    for (int k0 = 0; k0 < SEGK; k0 += 32) {
      v8h lo = *(const v8h*)(ap + k0);
      v8h hi = *(const v8h*)(ap + k0 + 8);
      v16h af = __builtin_shufflevector(lo, hi, 0, 1, 2, 3, 4, 5, 6, 7,
                                        8, 9, 10, 11, 12, 13, 14, 15);
      int kt = (s * SEGK + k0) >> 5;
#pragma unroll
      for (int t = 0; t < NT; ++t) {
        v16h bf = *(const v16h*)(blane + ((size_t)kt * ntiles + nt0 + t) * 512);
        acc[t] = __builtin_amdgcn_wmma_f32_16x16x32_f16(
            false, af, false, bf, (short)0, acc[t], false, false);
      }
    }
  }

  int rbase = row0 + ((lane >> 4) << 3);
#pragma unroll
  for (int t = 0; t < NT; ++t) {
    int n  = (nt0 + t) * 16 + (lane & 15);
    int nc = (n < N) ? n : (N - 1);
    float bv = bias ? bias[nc] : 0.f;
#pragma unroll
    for (int v = 0; v < 8; ++v) {
      int r  = rbase + v;
      int rr = (r < M) ? r : (M - 1);
      float x = acc[t][v] + bv;
      if (act) x = siluf(x);
      if (mulIn) {
        int mr = mulG ? mulG[rr] : rr;
        x *= (float)mulIn[(size_t)mr * mulLd + nc];
      }
      if (addIn) x += (float)addIn[(size_t)rr * addLd + nc];
      bool ok = (r < M) && (n < N);
      if (out16 && ok)   out16[(size_t)r * ldo + n] = (h16)x;
      if (scatOut && ok) atomicAdd(scatOut + (size_t)scatIdx[rr] * scatLd + n, x);
    }
  }
}

// ---------------------------------------------------------------------------
// Small helpers / VALU kernels
// ---------------------------------------------------------------------------
__global__ void zero_f32(float* p, int n) {
  int i = blockIdx.x * blockDim.x + threadIdx.x;
  if (i < n) p[i] = 0.f;
}

__global__ void cvt_f2h(const float* __restrict__ s, h16* __restrict__ d, int n) {
  int i = blockIdx.x * blockDim.x + threadIdx.x;
  if (i < n) d[i] = (h16)s[i];
}

// f32 (K x N) -> f16 fragment-swizzled (Kp x Np), zero-padded.
__global__ void pack_b(const float* __restrict__ src, h16* __restrict__ dst,
                       int K, int N, int Kp, int Np) {
  int i = blockIdx.x * blockDim.x + threadIdx.x;
  if (i >= Kp * Np) return;
  int j = i & 15;
  int chunk = i >> 4;
  int lane = chunk & 31;
  int tile = chunk >> 5;
  int ntiles = Np >> 4;
  int nt = tile % ntiles, kt = tile / ntiles;
  int k = kt * 32 + ((lane >> 4) << 4) + j;
  int n = nt * 16 + (lane & 15);
  dst[i] = (k < K && n < N) ? (h16)src[k * N + n] : (h16)0.f;
}

// fragment-swizzled f16 C[Kp x Np] = A[K x J](f32) @ B[J x N](f32), zero pad
__global__ void small_mm_packed(const float* __restrict__ A,
                                const float* __restrict__ B, h16* __restrict__ C,
                                int K, int J, int N, int Kp, int Np) {
  int i = blockIdx.x * blockDim.x + threadIdx.x;
  if (i >= Kp * Np) return;
  int j16 = i & 15;
  int chunk = i >> 4;
  int lane = chunk & 31;
  int tile = chunk >> 5;
  int ntiles = Np >> 4;
  int nt = tile % ntiles, kt = tile / ntiles;
  int k = kt * 32 + ((lane >> 4) << 4) + j16;
  int n = nt * 16 + (lane & 15);
  float s = 0.f;
  if (k < K && n < N)
    for (int j = 0; j < J; ++j) s += A[k * J + j] * B[j * N + n];
  C[i] = (h16)s;
}

// C[K x N](f32) = A[K x J](f32) @ B[J x N](f32)
__global__ void small_mm_f(const float* __restrict__ A,
                           const float* __restrict__ B, float* __restrict__ C,
                           int K, int J, int N) {
  int i = blockIdx.x * blockDim.x + threadIdx.x;
  if (i >= K * N) return;
  int k = i / N, n = i - k * N;
  float s = 0.f;
  for (int j = 0; j < J; ++j) s += A[k * J + j] * B[j * N + n];
  C[i] = s;
}

__global__ void init_tables(float* tab) {   // [0..41]=bessel norm, [42..48]=sph coef
  int i = threadIdx.x;
  if (i < 42) {
    int l = i / 6;
    float j = sphjn(c_bz[i], l + 1);
    tab[i] = rsqrtf(0.5f * j * j);
  } else if (i < 49) {
    int l = i - 42;
    tab[i] = sqrtf((2.f * l + 1.f) / (4.f * 3.14159265358979f));
  }
}

__global__ void embed_nodes(const float* __restrict__ atomW,
                            const int* __restrict__ Z, h16* __restrict__ h, int n) {
  int i = blockIdx.x * blockDim.x + threadIdx.x;
  if (i >= n) return;
  int node = i >> 7, c = i & 127;
  h[i] = (h16)atomW[Z[node] * 128 + c];
}

// Edge geometry: o, rbf (env*sin(freq*ds)), rbf_env (env*norm*j_l(ds*z))
__global__ void edge_kernel(const float* __restrict__ pos,
                            const int* __restrict__ src, const int* __restrict__ dst,
                            const float* __restrict__ freq,
                            const float* __restrict__ tab,
                            float* __restrict__ o, float* __restrict__ rbf,
                            h16* __restrict__ rbfenv, int E) {
  int e = blockIdx.x * blockDim.x + threadIdx.x;
  if (e >= E) return;
  int s = src[e], d = dst[e];
  float ox = pos[d * 3 + 0] - pos[s * 3 + 0];
  float oy = pos[d * 3 + 1] - pos[s * 3 + 1];
  float oz = pos[d * 3 + 2] - pos[s * 3 + 2];
  o[e * 3 + 0] = ox; o[e * 3 + 1] = oy; o[e * 3 + 2] = oz;
  float dd = sqrtf(ox * ox + oy * oy + oz * oz);
  float x  = dd * (1.f / 5.f);
  float x2 = x * x, x4 = x2 * x2, x5 = x4 * x;
  float env = 1.f / x - 28.f * x5 + 48.f * x5 * x - 21.f * x5 * x2;
  for (int n = 0; n < 6; ++n) rbf[e * 6 + n] = env * __sinf(freq[n] * x);
  for (int l = 0; l < 7; ++l)
    for (int n = 0; n < 6; ++n) {
      int i = l * 6 + n;
      rbfenv[(size_t)e * 42 + i] = (h16)(env * tab[i] * sphjn(x * c_bz[i], l));
    }
}

// Triplet angular basis: sbf[t, l*6+n] = rbf_env[lg_dst[t], l*6+n]*P_l(cos)*coef_l
__global__ void triplet_kernel(const float* __restrict__ o,
                               const int* __restrict__ lgs,
                               const int* __restrict__ lgd,
                               const h16* __restrict__ rbfenv,
                               const float* __restrict__ tab,
                               h16* __restrict__ sbf, int T) {
  int t = blockIdx.x * blockDim.x + threadIdx.x;
  if (t >= T) return;
  int a = lgs[t], b = lgd[t];
  float ax = o[a * 3], ay = o[a * 3 + 1], az = o[a * 3 + 2];
  float bx = o[b * 3], by = o[b * 3 + 1], bz = o[b * 3 + 2];
  float xd = ax * bx + ay * by + az * bz;
  float cx = ay * bz - az * by, cy = az * bx - ax * bz, cz = ax * by - ay * bx;
  float yc2 = cx * cx + cy * cy + cz * cz;
  float ct  = xd * rsqrtf(xd * xd + yc2 + 1e-20f);   // cos(atan2(|cross|, dot))
  float P[7]; P[0] = 1.f; P[1] = ct;
  for (int l = 2; l < 7; ++l)
    P[l] = ((2.f * l - 1.f) * ct * P[l - 1] - (l - 1.f) * P[l - 2]) / (float)l;
  h16* row = sbf + (size_t)t * 64;
  const h16* re = rbfenv + (size_t)b * 42;
  for (int l = 0; l < 7; ++l) {
    float cb = P[l] * tab[42 + l];
    for (int n = 0; n < 6; ++n) row[l * 6 + n] = (h16)((float)re[l * 6 + n] * cb);
  }
  for (int k = 42; k < 64; ++k) row[k] = (h16)0.f;   // zero K-padding
}

// out[e,c] = opt_silu( sum_j rbf[e,j]*W[j,N]+b ) : K=6 matvec (VALU path)
__global__ void rbf_matvec(const float* __restrict__ rbf,
                           const float* __restrict__ W,
                           const float* __restrict__ bias,
                           int act, h16* __restrict__ out, int N, int M) {
  int e = blockIdx.x, c = threadIdx.x;
  if (e >= M) return;
  float s = bias ? bias[c] : 0.f;
#pragma unroll
  for (int j = 0; j < 6; ++j) s += rbf[e * 6 + j] * W[j * N + c];
  if (act) s = siluf(s);
  out[(size_t)e * N + c] = (h16)s;
}

// node_acc[src[e],c] += m16[e,c] * (rbf[e,:] @ Wrbf[:,c])
__global__ void outblock_scatter(const float* __restrict__ rbf,
                                 const float* __restrict__ Wrbf,
                                 const h16* __restrict__ m,
                                 const int* __restrict__ src,
                                 float* nodeacc, int M) {
  int e = blockIdx.x, c = threadIdx.x;
  if (e >= M) return;
  float s = 0.f;
#pragma unroll
  for (int j = 0; j < 6; ++j) s += rbf[e * 6 + j] * Wrbf[j * 128 + c];
  float v = s * (float)m[(size_t)e * 128 + c];
  atomicAdd(nodeacc + (size_t)src[e] * 128 + c, v);
}

// ---------------------------------------------------------------------------
// Host orchestration
// ---------------------------------------------------------------------------
static void launch_gemm(hipStream_t st,
    const h16* A0, const int* g0, const h16* A1, const int* g1,
    const h16* A2, const int* g2, int nseg, int segK,
    const h16* Bp, int Np, const float* bias, int act,
    const h16* mulIn, const int* mulG, int mulLd,
    const h16* addIn, int addLd,
    h16* out16, int ldo,
    float* scat, const int* scatIdx, int scatLd,
    int M, int N)
{
  int NT = (Np % 32 == 0) ? 2 : 1;
  int ntiles = Np / 16;
  dim3 grid((M + 127) / 128, Np / (16 * NT));
#define GEMM_GO(NS, SK, T)                                                    \
  wmma_gemm<NS, SK, T><<<grid, 256, 0, st>>>(A0, A1, A2, g0, g1, g2, Bp,      \
      ntiles, bias, act, mulIn, mulG, mulLd, addIn, addLd, out16, ldo,        \
      scat, scatIdx, scatLd, M, N)
  if (nseg == 3 && segK == 128)      { GEMM_GO(3, 128, 2); }
  else if (nseg == 1 && segK == 128) { GEMM_GO(1, 128, 2); }
  else if (nseg == 1 && segK == 64)  { GEMM_GO(1, 64, 2); }
  else if (nseg == 1 && segK == 256) {
    if (NT == 2) { GEMM_GO(1, 256, 2); } else { GEMM_GO(1, 256, 1); }
  }
#undef GEMM_GO
}

extern "C" void kernel_launch(void* const* d_in, const int* in_sizes, int n_in,
                              void* d_out, int out_size, void* d_ws, size_t ws_size,
                              hipStream_t stream) {
  (void)in_sizes; (void)n_in; (void)out_size; (void)ws_size;
  const int E = N_EDGES_C, T = N_TRIP_C, NN = N_NODES_C;

  const int*   Z    = (const int*)d_in[0];
  const float* pos  = (const float*)d_in[1];
  const int*   src  = (const int*)d_in[2];
  const int*   dst  = (const int*)d_in[3];
  const int*   lgs  = (const int*)d_in[4];
  const int*   lgd  = (const int*)d_in[5];
  const int*   ngid = (const int*)d_in[6];
  auto pf = [&](int i) { return (const float*)d_in[i]; };
  // Param leaf order (jax tree flatten, dict keys sorted):
  // 7:emb.W 8:emb.atom 9:emb.b 10:emb.rbf_W 11:emb.rbf_b 12:freq
  // int[i] @ 13+24i: +0..7 after(W1,b1,W2,b2)x2, +8..11 before, +12 down,
  //   +13 fbs_W +14 fbs_b +15 ji_W +16 ji_b +17 kj_W +18 kj_b
  //   +19 rbf1 +20 rbf2 +21 sbf1 +22 sbf2 +23 up
  // out[j] @ 109+9j: +0..5 dense(W,b)x3, +6 final, +7 rbf, +8 up

  uintptr_t base = (uintptr_t)d_ws;
  size_t off = 0;
  auto alloc = [&](size_t bytes) -> void* {
    void* p = (void*)(base + off);
    off = (off + bytes + 255) & ~(size_t)255;
    return p;
  };

  float* tab     = (float*)alloc(49 * 4);
  float* o_vec   = (float*)alloc((size_t)E * 3 * 4);
  float* rbf     = (float*)alloc((size_t)E * 6 * 4);
  h16*   rbfenv  = (h16*)alloc((size_t)E * 42 * 2);
  h16*   hemb    = (h16*)alloc((size_t)NN * 128 * 2);
  h16*   m16     = (h16*)alloc((size_t)E * 128 * 2);
  h16*   xji     = (h16*)alloc((size_t)E * 128 * 2);
  h16*   rbft    = (h16*)alloc((size_t)E * 128 * 2);
  h16*   tmpE    = (h16*)alloc((size_t)E * 128 * 2);
  h16*   mupd    = (h16*)alloc((size_t)E * 128 * 2);
  h16*   xkj     = (h16*)alloc((size_t)E * 64 * 2);
  h16*   seg16   = (h16*)alloc((size_t)E * 64 * 2);
  float* seg32   = (float*)alloc((size_t)E * 64 * 4);
  h16*   sbf     = (h16*)alloc((size_t)T * 64 * 2);
  float* nodeacc = (float*)alloc((size_t)NN * 128 * 4);
  h16*   node16  = (h16*)alloc((size_t)NN * 128 * 2);
  h16*   ntA     = (h16*)alloc((size_t)NN * 256 * 2);
  h16*   ntB     = (h16*)alloc((size_t)NN * 256 * 2);

  // f16 weight arena (fragment-swizzled)
  h16* embW16 = (h16*)alloc(384 * 128 * 2);
  h16 *wji[4], *wkj[4], *wfbs[4], *wbW1[4], *wbW2[4], *wdown[4], *wup[4], *wsbfW[4];
  h16 *waW1[4][2], *waW2[4][2];
  float* wrbft[4];
  for (int i = 0; i < 4; ++i) {
    wji[i] = (h16*)alloc(128 * 128 * 2);  wkj[i] = (h16*)alloc(128 * 128 * 2);
    wfbs[i] = (h16*)alloc(128 * 128 * 2);
    wbW1[i] = (h16*)alloc(128 * 128 * 2); wbW2[i] = (h16*)alloc(128 * 128 * 2);
    for (int a = 0; a < 2; ++a) {
      waW1[i][a] = (h16*)alloc(128 * 128 * 2);
      waW2[i][a] = (h16*)alloc(128 * 128 * 2);
    }
    wdown[i] = (h16*)alloc(128 * 64 * 2); wup[i] = (h16*)alloc(64 * 128 * 2);
    wsbfW[i] = (h16*)alloc(64 * 64 * 2);  wrbft[i] = (float*)alloc(6 * 128 * 4);
  }
  h16 *oup[5], *oden[5][3], *ofin[5];
  for (int j = 0; j < 5; ++j) {
    oup[j] = (h16*)alloc(128 * 256 * 2);
    for (int k = 0; k < 3; ++k) oden[j][k] = (h16*)alloc(256 * 256 * 2);
    ofin[j] = (h16*)alloc(256 * 16 * 2);
  }

  auto zeros = [&](float* p, int n) {
    zero_f32<<<(n + 255) / 256, 256, 0, stream>>>(p, n);
  };
  auto cvtw = [&](int idx, h16* dp, int K, int N, int Kp, int Np) {
    int n = Kp * Np;
    pack_b<<<(n + 255) / 256, 256, 0, stream>>>(pf(idx), dp, K, N, Kp, Np);
  };

  // ---- prep: tables, weights ----
  zeros((float*)d_out, N_GRAPHS_C * 12);
  init_tables<<<1, 64, 0, stream>>>(tab);
  cvtw(7, embW16, 384, 128, 384, 128);
  for (int i = 0; i < 4; ++i) {
    int ib = 13 + 24 * i;
    for (int a = 0; a < 2; ++a) {
      cvtw(ib + a * 4 + 0, waW1[i][a], 128, 128, 128, 128);
      cvtw(ib + a * 4 + 2, waW2[i][a], 128, 128, 128, 128);
    }
    cvtw(ib + 8, wbW1[i], 128, 128, 128, 128);
    cvtw(ib + 10, wbW2[i], 128, 128, 128, 128);
    cvtw(ib + 12, wdown[i], 128, 64, 128, 64);
    cvtw(ib + 13, wfbs[i], 128, 128, 128, 128);
    cvtw(ib + 15, wji[i], 128, 128, 128, 128);
    cvtw(ib + 17, wkj[i], 128, 128, 128, 128);
    cvtw(ib + 23, wup[i], 64, 128, 64, 128);
    small_mm_packed<<<(64 * 64 + 255) / 256, 256, 0, stream>>>(
        pf(ib + 21), pf(ib + 22), wsbfW[i], 42, 8, 64, 64, 64);
    small_mm_f<<<(6 * 128 + 255) / 256, 256, 0, stream>>>(
        pf(ib + 19), pf(ib + 20), wrbft[i], 6, 8, 128);
  }
  for (int j = 0; j < 5; ++j) {
    int ob = 109 + 9 * j;
    for (int k = 0; k < 3; ++k) cvtw(ob + 2 * k, oden[j][k], 256, 256, 256, 256);
    cvtw(ob + 6, ofin[j], 256, 12, 256, 16);
    cvtw(ob + 8, oup[j], 128, 256, 128, 256);
  }

  // ---- geometry + embedding ----
  edge_kernel<<<(E + 255) / 256, 256, 0, stream>>>(pos, src, dst, pf(12), tab,
                                                   o_vec, rbf, rbfenv, E);
  embed_nodes<<<(NN * 128 + 255) / 256, 256, 0, stream>>>(pf(8), Z, hemb, NN * 128);
  rbf_matvec<<<E, 128, 0, stream>>>(rbf, pf(10), pf(11), 1, tmpE, 128, E); // rbf_e
  // m = silu([h[src] | h[dst] | rbf_e] @ emb.W + emb.b)  (3-segment gather GEMM)
  launch_gemm(stream, hemb, src, hemb, dst, tmpE, nullptr, 3, 128,
              embW16, 128, pf(9), 1, nullptr, nullptr, 0, nullptr, 0,
              m16, 128, nullptr, nullptr, 0, E, 128);
  triplet_kernel<<<(T + 255) / 256, 256, 0, stream>>>(o_vec, lgs, lgd, rbfenv,
                                                      tab, sbf, T);

  // ---- output block helper ----
  auto out_block = [&](int j) {
    int ob = 109 + 9 * j;
    zeros(nodeacc, NN * 128);
    outblock_scatter<<<E, 128, 0, stream>>>(rbf, pf(ob + 7), m16, src, nodeacc, E);
    cvt_f2h<<<(NN * 128 + 255) / 256, 256, 0, stream>>>(nodeacc, node16, NN * 128);
    launch_gemm(stream, node16, nullptr, nullptr, nullptr, nullptr, nullptr, 1, 128,
                oup[j], 256, nullptr, 0, nullptr, nullptr, 0, nullptr, 0,
                ntA, 256, nullptr, nullptr, 0, NN, 256);
    launch_gemm(stream, ntA, nullptr, nullptr, nullptr, nullptr, nullptr, 1, 256,
                oden[j][0], 256, pf(ob + 1), 1, nullptr, nullptr, 0, nullptr, 0,
                ntB, 256, nullptr, nullptr, 0, NN, 256);
    launch_gemm(stream, ntB, nullptr, nullptr, nullptr, nullptr, nullptr, 1, 256,
                oden[j][1], 256, pf(ob + 3), 1, nullptr, nullptr, 0, nullptr, 0,
                ntA, 256, nullptr, nullptr, 0, NN, 256);
    launch_gemm(stream, ntA, nullptr, nullptr, nullptr, nullptr, nullptr, 1, 256,
                oden[j][2], 256, pf(ob + 5), 1, nullptr, nullptr, 0, nullptr, 0,
                ntB, 256, nullptr, nullptr, 0, NN, 256);
    // final (N=12, pad 16) fused with per-graph segment sum into d_out
    launch_gemm(stream, ntB, nullptr, nullptr, nullptr, nullptr, nullptr, 1, 256,
                ofin[j], 16, nullptr, 0, nullptr, nullptr, 0, nullptr, 0,
                nullptr, 0, (float*)d_out, ngid, 12, NN, 12);
  };

  out_block(0);

  // ---- interaction blocks ----
  for (int i = 0; i < 4; ++i) {
    int ib = 13 + 24 * i;
    rbf_matvec<<<E, 128, 0, stream>>>(rbf, wrbft[i], nullptr, 0, rbft, 128, E);
    // x_ji = silu(m @ ji_W + ji_b)
    launch_gemm(stream, m16, nullptr, nullptr, nullptr, nullptr, nullptr, 1, 128,
                wji[i], 128, pf(ib + 16), 1, nullptr, nullptr, 0, nullptr, 0,
                xji, 128, nullptr, nullptr, 0, E, 128);
    // tmp = silu(m @ kj_W + kj_b) * rbf_t
    launch_gemm(stream, m16, nullptr, nullptr, nullptr, nullptr, nullptr, 1, 128,
                wkj[i], 128, pf(ib + 18), 1, rbft, nullptr, 128, nullptr, 0,
                tmpE, 128, nullptr, nullptr, 0, E, 128);
    // x_kj = silu(tmp @ down)
    launch_gemm(stream, tmpE, nullptr, nullptr, nullptr, nullptr, nullptr, 1, 128,
                wdown[i], 64, nullptr, 1, nullptr, nullptr, 0, nullptr, 0,
                xkj, 64, nullptr, nullptr, 0, E, 64);
    // seg[lg_dst] += (sbf @ Wsbf) * x_kj[lg_src]   (fused scatter epilogue)
    zeros(seg32, E * 64);
    launch_gemm(stream, sbf, nullptr, nullptr, nullptr, nullptr, nullptr, 1, 64,
                wsbfW[i], 64, nullptr, 0, xkj, lgs, 64, nullptr, 0,
                nullptr, 0, seg32, lgd, 64, T, 64);
    cvt_f2h<<<(E * 64 + 255) / 256, 256, 0, stream>>>(seg32, seg16, E * 64);
    // m_upd = silu(seg @ up) + x_ji
    launch_gemm(stream, seg16, nullptr, nullptr, nullptr, nullptr, nullptr, 1, 64,
                wup[i], 128, nullptr, 1, nullptr, nullptr, 0, xji, 128,
                mupd, 128, nullptr, nullptr, 0, E, 128);
    // before residual
    launch_gemm(stream, mupd, nullptr, nullptr, nullptr, nullptr, nullptr, 1, 128,
                wbW1[i], 128, pf(ib + 9), 1, nullptr, nullptr, 0, nullptr, 0,
                tmpE, 128, nullptr, nullptr, 0, E, 128);
    launch_gemm(stream, tmpE, nullptr, nullptr, nullptr, nullptr, nullptr, 1, 128,
                wbW2[i], 128, pf(ib + 11), 1, nullptr, nullptr, 0, mupd, 128,
                mupd, 128, nullptr, nullptr, 0, E, 128);
    // m = m + silu(m_upd @ fbs_W + fbs_b)
    launch_gemm(stream, mupd, nullptr, nullptr, nullptr, nullptr, nullptr, 1, 128,
                wfbs[i], 128, pf(ib + 14), 1, nullptr, nullptr, 0, m16, 128,
                m16, 128, nullptr, nullptr, 0, E, 128);
    // after residuals x2
    for (int a = 0; a < 2; ++a) {
      launch_gemm(stream, m16, nullptr, nullptr, nullptr, nullptr, nullptr, 1, 128,
                  waW1[i][a], 128, pf(ib + a * 4 + 1), 1, nullptr, nullptr, 0,
                  nullptr, 0, tmpE, 128, nullptr, nullptr, 0, E, 128);
      launch_gemm(stream, tmpE, nullptr, nullptr, nullptr, nullptr, nullptr, 1, 128,
                  waW2[i][a], 128, pf(ib + a * 4 + 3), 1, nullptr, nullptr, 0,
                  m16, 128, m16, 128, nullptr, nullptr, 0, E, 128);
    }
    out_block(i + 1);
  }
}
